// WaveletVQEncoder_65017214927380
// MI455X (gfx1250) — matrix operand order repaired and checked by reading, not compiled, not run on hardware
//
#include <hip/hip_runtime.h>
#include <hip/hip_bf16.h>
#include <math.h>

typedef _Float16 v16h __attribute__((ext_vector_type(16)));
typedef float    v8f  __attribute__((ext_vector_type(8)));
typedef float    v4f  __attribute__((ext_vector_type(4)));

#define DIM            128
#define KCODES         512
#define ROWS_PER_BLOCK 16
#define COMMIT_COST    0.25f

// ---------------------------------------------------------------------------
// Prep: S = sig + sig^T ;  F = E @ S (f32 + f16 copy) ; quad_e[k] = sum(F*E)
// One block per codebook row k (512 blocks x 128 threads).
// ---------------------------------------------------------------------------
__global__ __launch_bounds__(128) void vq_prep(
    const float* __restrict__ E, const float* __restrict__ sig,
    _Float16* __restrict__ F16, float* __restrict__ quad_e)
{
    __shared__ float erow[DIM];
    __shared__ float red[DIM];
    const int k = blockIdx.x;
    const int d = threadIdx.x;

    erow[d] = E[k * DIM + d];
    __syncthreads();

    float f = 0.0f;
    for (int j = 0; j < DIM; ++j)
        f = fmaf(erow[j], sig[j * DIM + d] + sig[d * DIM + j], f);

    F16[k * DIM + d] = (_Float16)f;
    red[d] = f * erow[d];
    __syncthreads();
    for (int s = 64; s > 0; s >>= 1) {
        if (d < s) red[d] += red[d + s];
        __syncthreads();
    }
    if (d == 0) quad_e[k] = red[0];
}

// ---------------------------------------------------------------------------
// Main: per block of 16 rows, WMMA GEMM x(16x128) @ F16^T(128x512) + argmin,
// then gather q = E[idx], write quantized + indices, per-block partial sums.
// 256 threads = 8 waves; wave w owns codes [64w, 64w+64) as 4 16x16 tiles.
// ---------------------------------------------------------------------------
__global__ __launch_bounds__(256) void vq_main(
    const float* __restrict__ x,        // [N,128]
    const float* __restrict__ w,        // [N]
    const float* __restrict__ E,        // [512,128] f32
    const _Float16* __restrict__ F16,   // [512,128] f16  (= E @ S)
    const float* __restrict__ quad_e,   // [512]
    float* __restrict__ out_q,          // [N,128]
    float* __restrict__ out_idx,        // [N] (indices as float)
    float* __restrict__ partial_sq,     // [nBlocks]
    float* __restrict__ partial_w,      // [nBlocks]
    unsigned* __restrict__ counts)      // [512]
{
    __shared__ __align__(16) float    xs32[ROWS_PER_BLOCK * DIM]; // 8 KB
    __shared__ __align__(16) _Float16 xs16[ROWS_PER_BLOCK * DIM]; // 4 KB
    __shared__ float wave_bv[8][ROWS_PER_BLOCK];
    __shared__ int   wave_bi[8][ROWS_PER_BLOCK];
    __shared__ int   fidx[ROWS_PER_BLOCK];
    __shared__ float red[256];

    const int tid  = threadIdx.x;
    const int wave = tid >> 5;
    const int lane = tid & 31;
    const size_t rowBase = (size_t)blockIdx.x * ROWS_PER_BLOCK;

    // Stage x tile (f32 -> LDS, f16 copy for WMMA A operand)
    {
        const float* src = x + rowBase * DIM;
        for (int i = tid; i < ROWS_PER_BLOCK * DIM; i += 256) {
            float v = src[i];
            xs32[i] = v;
            xs16[i] = (_Float16)v;
        }
    }
    __syncthreads();

    // --- WMMA: cross[row, code] over this wave's 64 codes ---
    const int halfSel   = (lane < 16) ? 0 : 1;  // selects K-halves per ISA layout
    const int rowInTile = lane & 15;            // A: row index this lane supplies
    const int codeCol   = lane & 15;            // B/C: column within 16-code tile
    const int codeBase  = wave * 64;

    v8f acc[4] = {};   // 4 code tiles, f32 C/D fragments
    for (int ks = 0; ks < 4; ++ks) {           // reduce over D in slices of 32
        const int kbase = ks * 32;
        v16h a;
        {
            // 16-bit A 16x32 layout: lane<16 -> K{0..7,16..23}, lane>=16 -> K{8..15,24..31}
            const _Float16* ar = xs16 + rowInTile * DIM + kbase + halfSel * 8;
            v4f* ap = (v4f*)&a;
            ap[0] = *(const v4f*)(ar);
            ap[1] = *(const v4f*)(ar + 16);
        }
        for (int t = 0; t < 4; ++t) {
            const int code = codeBase + t * 16 + codeCol;
            // 16-bit B 32x16 layout: lane<16 -> K{0..7,16..23}, lane>=16 -> K{8..15,24..31}
            const _Float16* br = F16 + (size_t)code * DIM + kbase + halfSel * 8;
            v16h b;
            v4f* bp = (v4f*)&b;
            bp[0] = *(const v4f*)(br);
            bp[1] = *(const v4f*)(br + 16);
            acc[t] = __builtin_amdgcn_wmma_f32_16x16x32_f16(
                false, a, false, b, (short)0, acc[t], false, false);
        }
    }

    // --- score = quad_e[code] - 2*cross ; per-row running argmin ---
    // C layout: VGPR j holds row j (lanes 0-15) / row j+8 (lanes 16-31), col = lane%16
    float bv[8];
    int   bi[8];
    for (int j = 0; j < 8; ++j) { bv[j] = 3.4e38f; bi[j] = 0x7fffffff; }
    for (int t = 0; t < 4; ++t) {
        const int   code = codeBase + t * 16 + codeCol;
        const float qe   = quad_e[code];
        for (int j = 0; j < 8; ++j) {
            float s = fmaf(-2.0f, acc[t][j], qe);
            if (s < bv[j] || (s == bv[j] && code < bi[j])) { bv[j] = s; bi[j] = code; }
        }
    }
    // reduce across the 16 lanes of each half (xor masks stay within halves)
    for (int m = 1; m <= 8; m <<= 1) {
        for (int j = 0; j < 8; ++j) {
            float ov = __shfl_xor(bv[j], m, 32);
            int   oi = __shfl_xor(bi[j], m, 32);
            if (ov < bv[j] || (ov == bv[j] && oi < bi[j])) { bv[j] = ov; bi[j] = oi; }
        }
    }
    if ((lane & 15) == 0) {
        const int rbase = (lane >= 16) ? 8 : 0;
        for (int j = 0; j < 8; ++j) {
            wave_bv[wave][j + rbase] = bv[j];
            wave_bi[wave][j + rbase] = bi[j];
        }
    }
    __syncthreads();

    // combine the 8 waves' code-ranges -> final index per row
    if (tid < ROWS_PER_BLOCK) {
        float v  = wave_bv[0][tid];
        int   i0 = wave_bi[0][tid];
        for (int wv = 1; wv < 8; ++wv) {
            float ov = wave_bv[wv][tid];
            int   oi = wave_bi[wv][tid];
            if (ov < v || (ov == v && oi < i0)) { v = ov; i0 = oi; }
        }
        fidx[tid] = i0;
        out_idx[rowBase + tid] = (float)i0;
        atomicAdd(&counts[i0], 1u);
    }
    if (tid == 0) {
        float sw = 0.0f;
        for (int r = 0; r < ROWS_PER_BLOCK; ++r) sw += w[rowBase + r];
        partial_w[blockIdx.x] = sw;
    }
    __syncthreads();

    // --- quantized = E[idx] (straight-through forward value), squared error ---
    float lsq = 0.0f;
    for (int i = tid; i < ROWS_PER_BLOCK * DIM; i += 256) {
        const int r = i >> 7, d = i & (DIM - 1);
        float q    = E[(size_t)fidx[r] * DIM + d];
        float diff = q - xs32[i];
        lsq = fmaf(diff, diff, lsq);
        __builtin_nontemporal_store(q, &out_q[rowBase * DIM + i]);
    }
    red[tid] = lsq;
    __syncthreads();
    for (int s = 128; s > 0; s >>= 1) {
        if (tid < s) red[tid] += red[tid + s];
        __syncthreads();
    }
    if (tid == 0) partial_sq[blockIdx.x] = red[0];
}

// ---------------------------------------------------------------------------
// Final: deterministic reduction of per-block partials -> 3 scalars.
// ---------------------------------------------------------------------------
__global__ __launch_bounds__(256) void vq_final(
    const float* __restrict__ partial_sq, const float* __restrict__ partial_w,
    const unsigned* __restrict__ counts, float* __restrict__ out,
    int nblocks, int Ntot, size_t off_cb, size_t off_cm, size_t off_ppl)
{
    __shared__ float rs[256], rw[256], rp[256];
    const int tid = threadIdx.x;
    float ssq = 0.0f, sw = 0.0f;
    for (int i = tid; i < nblocks; i += 256) { ssq += partial_sq[i]; sw += partial_w[i]; }
    float sp = 0.0f;
    for (int i = tid; i < KCODES; i += 256) {
        float p = (float)counts[i] / (float)Ntot;
        sp += p * logf(p + 1e-10f);
    }
    rs[tid] = ssq; rw[tid] = sw; rp[tid] = sp;
    __syncthreads();
    for (int s = 128; s > 0; s >>= 1) {
        if (tid < s) { rs[tid] += rs[tid + s]; rw[tid] += rw[tid + s]; rp[tid] += rp[tid + s]; }
        __syncthreads();
    }
    if (tid == 0) {
        float mse = rs[0] / ((float)Ntot * (float)DIM);
        out[off_cb]  = mse * (rw[0] / (float)Ntot);
        out[off_cm]  = mse * COMMIT_COST;
        out[off_ppl] = expf(-rp[0]);
    }
}

// ---------------------------------------------------------------------------
extern "C" void kernel_launch(void* const* d_in, const int* in_sizes, int n_in,
                              void* d_out, int out_size, void* d_ws, size_t ws_size,
                              hipStream_t stream) {
    const float* x   = (const float*)d_in[0];   // [B,C,T,D] f32
    const float* w   = (const float*)d_in[1];   // [B,C,T]   f32
    const float* E   = (const float*)d_in[2];   // [K,D]     f32
    const float* sig = (const float*)d_in[3];   // [D,D]     f32

    const int N       = in_sizes[0] / DIM;      // 131072 rows
    const int nBlocks = N / ROWS_PER_BLOCK;     // 8192

    // workspace layout (all offsets 256B-aligned)
    char* ws = (char*)d_ws;
    _Float16* F16       = (_Float16*)(ws);                          // 512*128*2 = 128 KB
    float*    quad_e    = (float*)(ws + 131072);                    // 2 KB
    float*    partial_sq= (float*)(ws + 131072 + 2048);             // 32 KB
    float*    partial_w = (float*)(ws + 131072 + 2048 + 32768);     // 32 KB
    unsigned* counts    = (unsigned*)(ws + 131072 + 2048 + 65536);  // 2 KB

    // d_out layout: quantized [N*D], codebook_loss, commitment_loss,
    //               encoding_indices [N] (as float), perplexity
    float* out = (float*)d_out;
    const size_t off_cb  = (size_t)N * DIM;
    const size_t off_cm  = off_cb + 1;
    const size_t off_idx = off_cm + 1;
    const size_t off_ppl = off_idx + (size_t)N;

    hipMemsetAsync(counts, 0, KCODES * sizeof(unsigned), stream);

    vq_prep<<<KCODES, DIM, 0, stream>>>(E, sig, F16, quad_e);

    vq_main<<<nBlocks, 256, 0, stream>>>(
        x, w, E, F16, quad_e,
        out /* quantized */, out + off_idx,
        partial_sq, partial_w, counts);

    vq_final<<<1, 256, 0, stream>>>(partial_sq, partial_w, counts, out,
                                    nBlocks, N, off_cb, off_cm, off_ppl);
}
